// MultiheadAttention_16372415333003
// MI455X (gfx1250) — matrix operand order repaired and checked by reading, not compiled, not run on hardware
//
// CDNA5 (gfx1250) single-head causal attention, bf16 WMMA pipeline.
// Pipeline: cvt weights -> Q/K/V projections (WMMA, V stored transposed)
//           -> softmax stats (online, WMMA scores, Q tile async-DMA'd to LDS)
//           -> per-chunk: materialize P (bf16) -> causal ctx GEMM (WMMA)
//           -> output projection (WMMA, f32 out + bias).
// gfx1250-specific paths: v_wmma_f32_16x16x32_bf16 everywhere,
// global_load_async_to_lds_b128 + s_wait_asynccnt (ASYNCcnt) for LDS staging,
// global_prefetch_b8 for the K stream.
#include <hip/hip_runtime.h>

#define DEV __device__ __forceinline__

typedef __attribute__((ext_vector_type(16))) __bf16 v16bf;
typedef __attribute__((ext_vector_type(8)))  __bf16 v8bf;
typedef __attribute__((ext_vector_type(8)))  float  v8f;

// ---- scalar helpers -------------------------------------------------------
DEV __bf16 f2bf(float f) {
  unsigned u = __builtin_bit_cast(unsigned, f);
  u += 0x7FFFu + ((u >> 16) & 1u);            // round-to-nearest-even
  unsigned short h = (unsigned short)(u >> 16);
  return __builtin_bit_cast(__bf16, h);
}

DEV v8f wmma_bf16(v16bf a, v16bf b, v8f c) {
  return __builtin_amdgcn_wmma_f32_16x16x32_bf16(false, a, false, b,
                                                 (short)0, c, false, false);
}

// A fragment (16x32, M x K), bf16 row-major source, base at [row0][k0].
// lane l: row = l%16; elems 0..7 -> K = 8*(l/16)+j ; 8..15 -> K = 16+8*(l/16)+j
DEV v16bf load_a_bf(const __bf16* base, int ld) {
  int lane = threadIdx.x & 31;
  const __bf16* p = base + (size_t)(lane & 15) * ld + 8 * (lane >> 4);
  v8bf lo = *(const v8bf*)p;
  v8bf hi = *(const v8bf*)(p + 16);
  return __builtin_shufflevector(lo, hi, 0,1,2,3,4,5,6,7,8,9,10,11,12,13,14,15);
}

// Same A fragment but converting from f32 source on the fly.
DEV v16bf load_a_f32(const float* base, int ld) {
  int lane = threadIdx.x & 31;
  const float* p = base + (size_t)(lane & 15) * ld + 8 * (lane >> 4);
  v16bf a;
#pragma unroll
  for (int i = 0; i < 8; ++i) { a[i] = f2bf(p[i]); a[i + 8] = f2bf(p[16 + i]); }
  return a;
}

// B fragment (32x16, K x N) where B[k][n] = W[n][k], W row-major [N][K].
// lane l: n = l%16; elems 0..7 -> K = 8*(l/16)+j ; 8..15 -> K = 16+8*(l/16)+j
DEV v16bf load_b_bf(const __bf16* base, int ld) {
  int lane = threadIdx.x & 31;
  const __bf16* p = base + (size_t)(lane & 15) * ld + 8 * (lane >> 4);
  v8bf lo = *(const v8bf*)p;
  v8bf hi = *(const v8bf*)(p + 16);
  return __builtin_shufflevector(lo, hi, 0,1,2,3,4,5,6,7,8,9,10,11,12,13,14,15);
}

// Async DMA one 64-query x 1024 bf16 tile (128 KB) from global into LDS.
// Uses GLOBAL_LOAD_ASYNC_TO_LDS_B128 (ASYNCcnt), then drains and barriers.
// Per wave: consecutive lanes cover contiguous 512B per issued instruction.
DEV void stage_tile_async(const __bf16* gsrc, __bf16* qs_shared) {
  unsigned lds0 = (unsigned)(uintptr_t)qs_shared;   // flat->LDS: low 32 bits
  const char* g = (const char*)gsrc;
  for (int i = threadIdx.x; i < (64 * 1024 * 2) / 16; i += 128) {
    asm volatile("global_load_async_to_lds_b128 %0, %1, off"
                 :: "v"(lds0 + (unsigned)i * 16u),
                    "v"(g + (size_t)i * 16)
                 : "memory");
  }
  asm volatile("s_wait_asynccnt 0" ::: "memory");
  __syncthreads();
}

// 16x32 score tile S = Qtile(16x1024) * Ktile(32x1024)^T via 64 WMMAs.
DEV void score_tile(const __bf16* qrow, const __bf16* krow, v8f& s0, v8f& s1) {
  for (int k = 0; k < 1024; k += 32) {
    v16bf a  = load_a_bf(qrow + k, 1024);
    v16bf b0 = load_b_bf(krow + k, 1024);
    v16bf b1 = load_b_bf(krow + 16 * 1024 + k, 1024);
    s0 = wmma_bf16(a, b0, s0);
    s1 = wmma_bf16(a, b1, s1);
  }
}

// ---- weight conversion ----------------------------------------------------
__global__ void cvt_bf16(const float* __restrict__ src, __bf16* __restrict__ dst, int n) {
  int i = blockIdx.x * 256 + threadIdx.x;
  if (i < n) dst[i] = f2bf(src[i]);
}

// ---- generic GEMM: out[m][n] = sum_k A[m][k]*W[n][k] + bias[n] ------------
// MODE 0: bf16 out row-major [M][1024]
// MODE 1: bf16 out transposed per batch: out[(b*1024+n)*4096 + t]
// MODE 2: f32  out row-major [M][1024]
template <int MODE, bool AF32>
__global__ __launch_bounds__(128) void proj_gemm(const void* __restrict__ Asrc,
    const __bf16* __restrict__ W, const float* __restrict__ bias,
    void* __restrict__ outp, int K)
{
  const int N = 1024;
  int lane = threadIdx.x & 31, wave = threadIdx.x >> 5;
  int m0 = blockIdx.y * 64 + wave * 16;
  int n0 = blockIdx.x * 64;
  v8f acc[4] = {{}, {}, {}, {}};
  for (int k = 0; k < K; k += 32) {
    v16bf a;
    if (AF32) a = load_a_f32((const float*)Asrc + (size_t)m0 * K + k, K);
    else      a = load_a_bf((const __bf16*)Asrc + (size_t)m0 * K + k, K);
#pragma unroll
    for (int f = 0; f < 4; ++f) {
      v16bf b = load_b_bf(W + (size_t)(n0 + 16 * f) * K + k, K);
      acc[f] = wmma_bf16(a, b, acc[f]);
    }
  }
  int n = lane & 15, h = lane >> 4;
#pragma unroll
  for (int f = 0; f < 4; ++f) {
    int col = n0 + 16 * f + n;
    float bv = bias[col];
    if (MODE == 0) {
#pragma unroll
      for (int v = 0; v < 8; ++v)
        ((__bf16*)outp)[(size_t)(m0 + 8 * h + v) * N + col] = f2bf(acc[f][v] + bv);
    } else if (MODE == 1) {
      int b  = m0 >> 12;
      int t0 = (m0 & 4095) + 8 * h;
      v8bf pk;
#pragma unroll
      for (int v = 0; v < 8; ++v) pk[v] = f2bf(acc[f][v] + bv);
      *(v8bf*)((__bf16*)outp + (((size_t)(b * N + col)) << 12) + t0) = pk;
    } else {
#pragma unroll
      for (int v = 0; v < 8; ++v)
        ((float*)outp)[(size_t)(m0 + 8 * h + v) * N + col] = acc[f][v] + bv;
    }
  }
}

// ---- pass 1: per-row softmax max/sum (online), causal ---------------------
__global__ __launch_bounds__(128) void attn_stats(const __bf16* __restrict__ Qb,
    const __bf16* __restrict__ Kb, float* __restrict__ ms, float* __restrict__ ls)
{
  __shared__ __bf16 qs[64 * 1024];          // 128 KB Q tile (64 queries)
  int lane = threadIdx.x & 31, wave = threadIdx.x >> 5;
  int qg = blockIdx.x * 64;                 // global row base [0, B*T)
  int b  = qg >> 12;
  stage_tile_async(Qb + (size_t)qg * 1024, qs);
  int qw = (qg & 4095) + wave * 16;         // first query of this wave (in batch)
  const __bf16* qrow  = qs + wave * 16 * 1024;
  const __bf16* kbase = Kb + ((size_t)b << 12) * 1024;
  int h = lane >> 4, n = lane & 15;
  float rm[8], rl[8];
#pragma unroll
  for (int v = 0; v < 8; ++v) { rm[v] = -3.0e38f; rl[v] = 0.f; }
  for (int t0 = 0; t0 <= qw + 15; t0 += 32) {
    __builtin_prefetch(kbase + (size_t)(t0 + 32) * 1024, 0, 1);
    v8f s0 = {}, s1 = {};
    score_tile(qrow, kbase + (size_t)t0 * 1024, s0, s1);
#pragma unroll
    for (int v = 0; v < 8; ++v) {
      int q = qw + 8 * h + v;
      float x0 = (t0 + n      <= q) ? s0[v] * 0.03125f : -3.0e38f;
      float x1 = (t0 + 16 + n <= q) ? s1[v] * 0.03125f : -3.0e38f;
      float mx = fmaxf(x0, x1);
#pragma unroll
      for (int d = 1; d < 16; d <<= 1) mx = fmaxf(mx, __shfl_xor(mx, d, 16));
      float nm = fmaxf(rm[v], mx);
      float es = __expf(x0 - nm) + __expf(x1 - nm);
#pragma unroll
      for (int d = 1; d < 16; d <<= 1) es += __shfl_xor(es, d, 16);
      rl[v] = rl[v] * __expf(rm[v] - nm) + es;
      rm[v] = nm;
    }
  }
  if (n == 0) {
#pragma unroll
    for (int v = 0; v < 8; ++v) {
      ms[qg + wave * 16 + 8 * h + v] = rm[v];
      ls[qg + wave * 16 + 8 * h + v] = rl[v];
    }
  }
}

// ---- pass 2a: materialize P chunk (1024 queries/batch) as bf16 ------------
__global__ __launch_bounds__(128) void attn_pk(const __bf16* __restrict__ Qb,
    const __bf16* __restrict__ Kb, const float* __restrict__ ms,
    const float* __restrict__ ls, __bf16* __restrict__ P, int qc0)
{
  __shared__ __bf16 qs[64 * 1024];
  int lane = threadIdx.x & 31, wave = threadIdx.x >> 5;
  int b   = blockIdx.x >> 4;                 // QC/64 = 16 blocks per batch
  int qq  = (blockIdx.x & 15) * 64;          // base within chunk
  int qib = qc0 + qq;                        // base within batch
  size_t qg = ((size_t)b << 12) + qib;       // global row base
  stage_tile_async(Qb + qg * 1024, qs);
  int qw = qib + wave * 16;
  const __bf16* qrow  = qs + wave * 16 * 1024;
  const __bf16* kbase = Kb + ((size_t)b << 12) * 1024;
  int h = lane >> 4, n = lane & 15;
  float mrow[8], li[8];
#pragma unroll
  for (int v = 0; v < 8; ++v) {
    int gr = (int)qg + wave * 16 + 8 * h + v;
    mrow[v] = ms[gr];
    li[v]   = 1.0f / ls[gr];
  }
  size_t prow0 = (size_t)(b * 1024 + qq + wave * 16);   // chunk-local P row
  for (int t0 = 0; t0 <= qw + 15; t0 += 32) {
    __builtin_prefetch(kbase + (size_t)(t0 + 32) * 1024, 0, 1);
    v8f s0 = {}, s1 = {};
    score_tile(qrow, kbase + (size_t)t0 * 1024, s0, s1);
#pragma unroll
    for (int v = 0; v < 8; ++v) {
      int q = qw + 8 * h + v;
      float x0 = (t0 + n      <= q) ? s0[v] * 0.03125f : -3.0e38f;
      float x1 = (t0 + 16 + n <= q) ? s1[v] * 0.03125f : -3.0e38f;
      __bf16* pr = P + ((prow0 + 8 * h + v) << 12);
      pr[t0 + n]      = f2bf(__expf(x0 - mrow[v]) * li[v]);   // masked -> 0
      pr[t0 + 16 + n] = f2bf(__expf(x1 - mrow[v]) * li[v]);
    }
  }
}

// ---- pass 2b: ctx = P * V  (causal K-bound per wave; V pre-transposed) ----
__global__ __launch_bounds__(128) void ctx_gemm(const __bf16* __restrict__ P,
    const __bf16* __restrict__ Vt, __bf16* __restrict__ ctx, int qc0)
{
  int lane = threadIdx.x & 31, wave = threadIdx.x >> 5;
  int mrow0 = blockIdx.y * 64 + wave * 16;   // chunk row [0, 4096)
  int b  = mrow0 >> 10;
  int qw = qc0 + (mrow0 & 1023);             // query within batch
  int n0 = blockIdx.x * 64;                  // output-feature slice
  v8f acc[4] = {{}, {}, {}, {}};
  const __bf16* arow  = P + ((size_t)mrow0 << 12);
  const __bf16* vbase = Vt + ((size_t)b << 22);
  int ktiles = ((qw + 15) >> 5) + 1;         // only tiles with t0 <= qw+15
  for (int kt = 0; kt < ktiles; ++kt) {
    int k = kt << 5;
    v16bf a = load_a_bf(arow + k, 4096);
#pragma unroll
    for (int f = 0; f < 4; ++f) {
      v16bf bb = load_b_bf(vbase + ((size_t)(n0 + 16 * f) << 12) + k, 4096);
      acc[f] = wmma_bf16(a, bb, acc[f]);
    }
  }
  int h = lane >> 4, n = lane & 15;
#pragma unroll
  for (int f = 0; f < 4; ++f)
#pragma unroll
    for (int v = 0; v < 8; ++v)
      ctx[(size_t)((b << 12) + qw + 8 * h + v) * 1024 + n0 + 16 * f + n] =
          f2bf(acc[f][v]);
}

// ---- host side ------------------------------------------------------------
extern "C" void kernel_launch(void* const* d_in, const int* in_sizes, int n_in,
                              void* d_out, int out_size, void* d_ws, size_t ws_size,
                              hipStream_t stream) {
  const float* x  = (const float*)d_in[0];
  const float* Wk = (const float*)d_in[1];
  const float* bk = (const float*)d_in[2];
  const float* Wq = (const float*)d_in[3];
  const float* bq = (const float*)d_in[4];
  const float* Wv = (const float*)d_in[5];
  const float* bv = (const float*)d_in[6];
  const float* Wo = (const float*)d_in[7];
  const float* bo = (const float*)d_in[8];
  float* out = (float*)d_out;

  char* ws = (char*)d_ws;
  const size_t SZ_QKV = (size_t)16384 * 1024 * 2;   // 32 MB each
  __bf16* Qb   = (__bf16*)(ws);
  __bf16* Kb   = (__bf16*)(ws + SZ_QKV);
  __bf16* Vtb  = (__bf16*)(ws + 2 * SZ_QKV);        // [B][1024][4096]
  __bf16* Ctxb = (__bf16*)(ws + 3 * SZ_QKV);
  __bf16* Pbuf = (__bf16*)(ws + 4 * SZ_QKV);        // [B][1024][4096] chunk
  __bf16* Wqb  = (__bf16*)(ws + 5 * SZ_QKV);
  __bf16* Wkb  = Wqb + 1024 * 1024;
  __bf16* Wvb  = Wkb + 1024 * 1024;
  __bf16* Wob  = Wvb + 1024 * 1024;
  float*  ms   = (float*)(Wob + 1024 * 1024);
  float*  ls   = ms + 16384;

  // 1) weights -> bf16
  cvt_bf16<<<4096, 256, 0, stream>>>(Wq, Wqb, 1024 * 1024);
  cvt_bf16<<<4096, 256, 0, stream>>>(Wk, Wkb, 1024 * 1024);
  cvt_bf16<<<4096, 256, 0, stream>>>(Wv, Wvb, 1024 * 1024);
  cvt_bf16<<<4096, 256, 0, stream>>>(Wo, Wob, 1024 * 1024);

  // 2) projections (A = x in f32, converted on the fly)
  dim3 gproj(16, 256);
  proj_gemm<0, true><<<gproj, 128, 0, stream>>>(x, Wqb, bq, Qb, 1024);
  proj_gemm<0, true><<<gproj, 128, 0, stream>>>(x, Wkb, bk, Kb, 1024);
  proj_gemm<1, true><<<gproj, 128, 0, stream>>>(x, Wvb, bv, Vtb, 1024);

  // 3) softmax statistics
  attn_stats<<<256, 128, 0, stream>>>(Qb, Kb, ms, ls);

  // 4) attention in 4 chunks of 1024 queries (per batch)
  for (int c = 0; c < 4; ++c) {
    attn_pk<<<64, 128, 0, stream>>>(Qb, Kb, ms, ls, Pbuf, c * 1024);
    ctx_gemm<<<dim3(16, 64), 128, 0, stream>>>(Pbuf, Vtb, Ctxb, c * 1024);
  }

  // 5) output projection -> f32 d_out (+ bias)
  proj_gemm<2, false><<<gproj, 128, 0, stream>>>(Ctxb, Wob, bo, out, 1024);
}
